// MultiScaleMaskedTransformerDecoder_35656818491969
// MI455X (gfx1250) — compile-verified, matrix operand order checked by reading, not run on hardware
//
#include <hip/hip_runtime.h>
#include <hip/hip_bf16.h>

typedef __attribute__((ext_vector_type(16))) __bf16 v16bf;
typedef __attribute__((ext_vector_type(8)))  float  v8f;

// ---------------- problem constants ----------------
constexpr int Bn  = 4;      // batch
constexpr int Dn  = 1024;   // embed
constexpr int Hn  = 8;      // heads
constexpr int HDn = 128;    // head dim
constexpr int FFn = 2048;
constexpr int NLn = 9;
constexpr int Sn  = 1024;   // 32*32 tokens
constexpr int BHn = Bn * Hn;

// ---------------- GEMM tiling ----------------
constexpr int BM = 128, BN = 128, BK = 32;
constexpr int LDT = BK + 16;   // 48 bf16 = 96B rows: keeps every b128 LDS access 16B-aligned

// ---- CDNA5 async global->LDS (ASYNCcnt) ----
static __device__ __forceinline__ unsigned lds_addr(const void* p) {
  return (unsigned)(unsigned long long)p;     // low 32 bits of generic LDS address = LDS offset
}
static __device__ __forceinline__ void async_b128(unsigned lds, const void* g) {
  unsigned long long ga = (unsigned long long)g;
  asm volatile("global_load_async_to_lds_b128 %0, %1, off"
               :: "v"(lds), "v"(ga) : "memory");
}
static __device__ __forceinline__ void wait_async0() {
  asm volatile("s_wait_asynccnt 0x0" ::: "memory");
}

static __device__ __forceinline__ v16bf load_frag(const __bf16* t) {
  // 16-bit A/B fragment layout (CDNA5 ISA 7.12.2):
  // lane&15 = row, lane>>4 selects K-base (0/8); elems 0..7 -> kb..kb+7, 8..15 -> kb+16..kb+23
  int lane = threadIdx.x & 31;
  const __bf16* p = t + (lane & 15) * LDT + ((lane >> 4) << 3);
  v16bf f;
#pragma unroll
  for (int e = 0; e < 8; ++e) f[e] = p[e];
#pragma unroll
  for (int e = 0; e < 8; ++e) f[8 + e] = p[16 + e];
  return f;
}
#define WMMA_BF16(a, b, c) \
  __builtin_amdgcn_wmma_f32_16x16x32_bf16(false, a, false, b, (short)0, c, false, false)

// ---------------- GEMM: C = A[M,K] * B[N,K]^T (+bias)(+relu), batched ----------------
__global__ __launch_bounds__(256)
void gemm_nt_kernel(const __bf16* __restrict__ A, const __bf16* __restrict__ Bm,
                    float* __restrict__ C, const float* __restrict__ bias,
                    int M, int N, int K, long sA, long sB, long sC, int act)
{
  __shared__ __bf16 As[2][BM * LDT];
  __shared__ __bf16 Bs[2][BN * LDT];
  const int z = blockIdx.z;
  A += (long)z * sA; Bm += (long)z * sB; C += (long)z * sC;
  const int tid  = threadIdx.x;
  const int m0   = blockIdx.y * BM, n0 = blockIdx.x * BN;
  const int wave = tid >> 5, lane = tid & 31;
  const int wm   = (wave >> 1) << 5;          // 4 wave-rows of 32
  const int wn   = (wave & 1) << 6;           // 2 wave-cols of 64
  v8f acc[2][4] = {};

  auto issue = [&](int k0, int buf) {
#pragma unroll
    for (int p = 0; p < 2; ++p) {             // A tile 128x32 -> 512 x b128
      int idx = p * 256 + tid;
      int r = idx >> 2, seg = (idx & 3) << 3;
      async_b128(lds_addr(&As[buf][r * LDT + seg]), A + (long)(m0 + r) * K + k0 + seg);
    }
#pragma unroll
    for (int p = 0; p < 2; ++p) {             // B tile 128x32
      int idx = p * 256 + tid;
      int r = idx >> 2, seg = (idx & 3) << 3;
      async_b128(lds_addr(&Bs[buf][r * LDT + seg]), Bm + (long)(n0 + r) * K + k0 + seg);
    }
  };

  issue(0, 0);
  const int nk = K / BK;
  for (int i = 0; i < nk; ++i) {
    const int cur = i & 1;
    wait_async0();
    __syncthreads();                           // buf[cur] published
    if (i + 1 < nk) issue((i + 1) * BK, cur ^ 1);
    v16bf a0 = load_frag(&As[cur][(wm +  0) * LDT]);
    v16bf a1 = load_frag(&As[cur][(wm + 16) * LDT]);
    v16bf b0 = load_frag(&Bs[cur][(wn +  0) * LDT]);
    v16bf b1 = load_frag(&Bs[cur][(wn + 16) * LDT]);
    v16bf b2 = load_frag(&Bs[cur][(wn + 32) * LDT]);
    v16bf b3 = load_frag(&Bs[cur][(wn + 48) * LDT]);
    acc[0][0] = WMMA_BF16(a0, b0, acc[0][0]);
    acc[0][1] = WMMA_BF16(a0, b1, acc[0][1]);
    acc[0][2] = WMMA_BF16(a0, b2, acc[0][2]);
    acc[0][3] = WMMA_BF16(a0, b3, acc[0][3]);
    acc[1][0] = WMMA_BF16(a1, b0, acc[1][0]);
    acc[1][1] = WMMA_BF16(a1, b1, acc[1][1]);
    acc[1][2] = WMMA_BF16(a1, b2, acc[1][2]);
    acc[1][3] = WMMA_BF16(a1, b3, acc[1][3]);
  }
  const int lr = (lane >> 4) << 3, lc = lane & 15;
#pragma unroll
  for (int i = 0; i < 2; ++i)
#pragma unroll
    for (int j = 0; j < 4; ++j) {
      int col = n0 + wn + j * 16 + lc;
      float bv = bias ? bias[col] : 0.f;
#pragma unroll
      for (int r = 0; r < 8; ++r) {
        int row = m0 + wm + i * 16 + lr + r;
        float v = acc[i][j][r] + bv;
        if (act) v = fmaxf(v, 0.f);
        C[(long)row * N + col] = v;
      }
    }
}

// ---------------- GEMM: C = A[M,K] * B[K,N], batched (att @ V); N == 128 ----------------
__global__ __launch_bounds__(256)
void gemm_nn_kernel(const __bf16* __restrict__ A, const __bf16* __restrict__ Bm,
                    float* __restrict__ C, int M, int N, int K,
                    long sA, long sB, long sC)
{
  __shared__ __bf16 As[2][BM * LDT];
  __shared__ __bf16 Bs[2][BN * LDT];
  const int z = blockIdx.z;
  A += (long)z * sA; Bm += (long)z * sB; C += (long)z * sC;
  const int tid  = threadIdx.x;
  const int m0   = blockIdx.y * BM, n0 = blockIdx.x * BN;
  const int wave = tid >> 5, lane = tid & 31;
  const int wm   = (wave >> 1) << 5, wn = (wave & 1) << 6;
  v8f acc[2][4] = {};

  auto issueA = [&](int k0, int buf) {
#pragma unroll
    for (int p = 0; p < 2; ++p) {
      int idx = p * 256 + tid;
      int r = idx >> 2, seg = (idx & 3) << 3;
      async_b128(lds_addr(&As[buf][r * LDT + seg]), A + (long)(m0 + r) * K + k0 + seg);
    }
  };
  auto loadB = [&](int k0, uint4* regs) {      // [K,N] source, vectorized over n
#pragma unroll
    for (int p = 0; p < 2; ++p) {
      int idx = p * 256 + tid;
      int c = idx >> 4, n8 = (idx & 15) << 3;
      regs[p] = *(const uint4*)(Bm + (long)(k0 + c) * N + n0 + n8);
    }
  };
  auto storeB = [&](int buf, const uint4* regs) {   // transpose into Bs[n][k]
#pragma unroll
    for (int p = 0; p < 2; ++p) {
      int idx = p * 256 + tid;
      int c = idx >> 4, n8 = (idx & 15) << 3;
      union { uint4 u; __bf16 h[8]; } ub; ub.u = regs[p];
#pragma unroll
      for (int j = 0; j < 8; ++j) Bs[buf][(n8 + j) * LDT + c] = ub.h[j];
    }
  };

  uint4 br[2];
  issueA(0, 0); loadB(0, br); storeB(0, br);
  const int nk = K / BK;
  for (int i = 0; i < nk; ++i) {
    const int cur = i & 1;
    wait_async0();
    __syncthreads();
    uint4 bnx[2];
    if (i + 1 < nk) { issueA((i + 1) * BK, cur ^ 1); loadB((i + 1) * BK, bnx); }
    v16bf a0 = load_frag(&As[cur][(wm +  0) * LDT]);
    v16bf a1 = load_frag(&As[cur][(wm + 16) * LDT]);
    v16bf b0 = load_frag(&Bs[cur][(wn +  0) * LDT]);
    v16bf b1 = load_frag(&Bs[cur][(wn + 16) * LDT]);
    v16bf b2 = load_frag(&Bs[cur][(wn + 32) * LDT]);
    v16bf b3 = load_frag(&Bs[cur][(wn + 48) * LDT]);
    acc[0][0] = WMMA_BF16(a0, b0, acc[0][0]);
    acc[0][1] = WMMA_BF16(a0, b1, acc[0][1]);
    acc[0][2] = WMMA_BF16(a0, b2, acc[0][2]);
    acc[0][3] = WMMA_BF16(a0, b3, acc[0][3]);
    acc[1][0] = WMMA_BF16(a1, b0, acc[1][0]);
    acc[1][1] = WMMA_BF16(a1, b1, acc[1][1]);
    acc[1][2] = WMMA_BF16(a1, b2, acc[1][2]);
    acc[1][3] = WMMA_BF16(a1, b3, acc[1][3]);
    if (i + 1 < nk) storeB(cur ^ 1, bnx);      // after compute: hides B load latency
  }
  const int lr = (lane >> 4) << 3, lc = lane & 15;
#pragma unroll
  for (int i = 0; i < 2; ++i)
#pragma unroll
    for (int j = 0; j < 4; ++j) {
      int col = n0 + wn + j * 16 + lc;
#pragma unroll
      for (int r = 0; r < 8; ++r) {
        int row = m0 + wm + i * 16 + lr + r;
        C[(long)row * N + col] = acc[i][j][r];
      }
    }
}

// ---------------- implicit-GEMM 3x3 conv tap over padded-NHWC input ----------------
// In: [B][34][34][3072] bf16 (halo zeros), W: [N=1024][K=3072] bf16 (per-tap pack)
__global__ __launch_bounds__(256)
void gemm_conv_tap_kernel(const __bf16* __restrict__ In, const __bf16* __restrict__ W,
                          float* __restrict__ C, int dy, int dx, int accum)
{
  const int N = 1024, K = 3072;
  __shared__ __bf16 As[2][BM * LDT];
  __shared__ __bf16 Bs[2][BN * LDT];
  const int tid  = threadIdx.x;
  const int m0   = blockIdx.y * BM, n0 = blockIdx.x * BN;
  const int wave = tid >> 5, lane = tid & 31;
  const int wm   = (wave >> 1) << 5, wn = (wave & 1) << 6;
  v8f acc[2][4] = {};

  auto issue = [&](int k0, int buf) {
#pragma unroll
    for (int p = 0; p < 2; ++p) {             // A: K-contiguous thanks to padded NHWC
      int idx = p * 256 + tid;
      int r = idx >> 2, seg = (idx & 3) << 3;
      int m = m0 + r;
      int bb = m >> 10, s = m & 1023;
      int py = (s >> 5) + dy + 1, px = (s & 31) + dx + 1;
      const __bf16* g = In + ((long)(bb * 34 + py) * 34 + px) * 3072 + k0 + seg;
      async_b128(lds_addr(&As[buf][r * LDT + seg]), g);
    }
#pragma unroll
    for (int p = 0; p < 2; ++p) {
      int idx = p * 256 + tid;
      int r = idx >> 2, seg = (idx & 3) << 3;
      async_b128(lds_addr(&Bs[buf][r * LDT + seg]), W + (long)(n0 + r) * K + k0 + seg);
    }
  };

  issue(0, 0);
  const int nk = K / BK;
  for (int i = 0; i < nk; ++i) {
    const int cur = i & 1;
    wait_async0();
    __syncthreads();
    if (i + 1 < nk) issue((i + 1) * BK, cur ^ 1);
    v16bf a0 = load_frag(&As[cur][(wm +  0) * LDT]);
    v16bf a1 = load_frag(&As[cur][(wm + 16) * LDT]);
    v16bf b0 = load_frag(&Bs[cur][(wn +  0) * LDT]);
    v16bf b1 = load_frag(&Bs[cur][(wn + 16) * LDT]);
    v16bf b2 = load_frag(&Bs[cur][(wn + 32) * LDT]);
    v16bf b3 = load_frag(&Bs[cur][(wn + 48) * LDT]);
    acc[0][0] = WMMA_BF16(a0, b0, acc[0][0]);
    acc[0][1] = WMMA_BF16(a0, b1, acc[0][1]);
    acc[0][2] = WMMA_BF16(a0, b2, acc[0][2]);
    acc[0][3] = WMMA_BF16(a0, b3, acc[0][3]);
    acc[1][0] = WMMA_BF16(a1, b0, acc[1][0]);
    acc[1][1] = WMMA_BF16(a1, b1, acc[1][1]);
    acc[1][2] = WMMA_BF16(a1, b2, acc[1][2]);
    acc[1][3] = WMMA_BF16(a1, b3, acc[1][3]);
  }
  const int lr = (lane >> 4) << 3, lc = lane & 15;
#pragma unroll
  for (int i = 0; i < 2; ++i)
#pragma unroll
    for (int j = 0; j < 4; ++j) {
      int col = n0 + wn + j * 16 + lc;
#pragma unroll
      for (int r = 0; r < 8; ++r) {
        int row = m0 + wm + i * 16 + lr + r;
        long off = (long)row * N + col;
        float v = acc[i][j][r];
        if (accum) v += C[off];
        C[off] = v;
      }
    }
}

// ---------------- elementwise / reduction kernels ----------------
__global__ void f2b_kernel(const float* __restrict__ x, __bf16* __restrict__ y, long n) {
  long i = (long)blockIdx.x * 256 + threadIdx.x;
  if (i < n) y[i] = (__bf16)x[i];
}
__global__ void zero_kernel(float* __restrict__ x, long n) {
  long i = (long)blockIdx.x * 256 + threadIdx.x;
  if (i < n) x[i] = 0.f;
}
// [B,D,S] NCHW -> [B,S,D] (f32 optional) + bf16
__global__ void transpose_feat_kernel(const float* __restrict__ in, float* __restrict__ oF,
                                      __bf16* __restrict__ oB) {
  long idx = (long)blockIdx.x * 256 + threadIdx.x;
  if (idx >= (long)Bn * Dn * Sn) return;
  int s = (int)(idx & 1023), d = (int)((idx >> 10) & 1023), b = (int)(idx >> 20);
  float v = in[idx];
  long o = ((long)b * Sn + s) * Dn + d;
  if (oF) oF[o] = v;
  oB[o] = (__bf16)v;
}
// y[b,s,d] = a[b,s,d] + q_nchw[b,d,s]
__global__ void add_nchw_kernel(const float* __restrict__ a, const float* __restrict__ q,
                                float* __restrict__ y) {
  long idx = (long)blockIdx.x * 256 + threadIdx.x;
  if (idx >= (long)Bn * Sn * Dn) return;
  int d = (int)(idx & 1023), s = (int)((idx >> 10) & 1023), b = (int)(idx >> 20);
  y[idx] = a[idx] + q[((long)b * Dn + d) * Sn + s];
}
// proj output [B,L,H,HD] f32 -> [(b*H+h),L,HD] bf16
__global__ void repack_to_bh_kernel(const float* __restrict__ in, __bf16* __restrict__ out) {
  long idx = (long)blockIdx.x * 256 + threadIdx.x;
  if (idx >= (long)Bn * Sn * Dn) return;
  int d = (int)(idx & 127), l = (int)((idx >> 7) & 1023);
  int h = (int)((idx >> 17) & 7), b = (int)(idx >> 20);
  out[idx] = (__bf16)in[((long)(b * Sn + l) * Hn + h) * HDn + d];
}
// ctx [(b*H+h),L,HD] f32 -> [B,L,D] bf16
__global__ void repack_from_bh_kernel(const float* __restrict__ in, __bf16* __restrict__ out) {
  long idx = (long)blockIdx.x * 256 + threadIdx.x;
  if (idx >= (long)Bn * Sn * Dn) return;
  int d2 = (int)(idx & 1023), l = (int)((idx >> 10) & 1023), b = (int)(idx >> 20);
  int h = d2 >> 7, d = d2 & 127;
  out[idx] = (__bf16)in[((long)(b * Hn + h) * Sn + l) * HDn + d];
}
__global__ __launch_bounds__(256)
void softmax_rows_kernel(float* __restrict__ X, __bf16* __restrict__ outB, int L, float scale) {
  const long row = blockIdx.x;
  float* x = X + row * (long)L;
  const int tid = threadIdx.x;
  __shared__ float red[256];
  float mx = -3.4e38f;
  for (int i = tid; i < L; i += 256) mx = fmaxf(mx, x[i]);
  red[tid] = mx; __syncthreads();
  for (int t = 128; t > 0; t >>= 1) { if (tid < t) red[tid] = fmaxf(red[tid], red[tid + t]); __syncthreads(); }
  mx = red[0]; __syncthreads();
  float sum = 0.f;
  for (int i = tid; i < L; i += 256) { float e = __expf((x[i] - mx) * scale); x[i] = e; sum += e; }
  red[tid] = sum; __syncthreads();
  for (int t = 128; t > 0; t >>= 1) { if (tid < t) red[tid] += red[tid + t]; __syncthreads(); }
  float inv = 1.f / red[0];
  for (int i = tid; i < L; i += 256) {
    float v = x[i] * inv; x[i] = v;
    if (outB) outB[row * (long)L + i] = (__bf16)v;
  }
}
__global__ __launch_bounds__(256)
void ln_rows_kernel(const float* __restrict__ A, const float* __restrict__ B2,
                    const float* __restrict__ g, const float* __restrict__ be,
                    float* __restrict__ outF, __bf16* __restrict__ outB) {
  const long row = blockIdx.x;
  const int tid = threadIdx.x;
  const float* a = A + row * Dn;
  float v[4]; float s = 0.f;
#pragma unroll
  for (int j = 0; j < 4; ++j) {
    int dcol = j * 256 + tid;
    float x = a[dcol] + (B2 ? B2[row * Dn + dcol] : 0.f);
    v[j] = x; s += x;
  }
  __shared__ float red[256];
  red[tid] = s; __syncthreads();
  for (int t = 128; t > 0; t >>= 1) { if (tid < t) red[tid] += red[tid + t]; __syncthreads(); }
  float mean = red[0] * (1.f / Dn); __syncthreads();
  float s2 = 0.f;
#pragma unroll
  for (int j = 0; j < 4; ++j) { float d = v[j] - mean; s2 += d * d; }
  red[tid] = s2; __syncthreads();
  for (int t = 128; t > 0; t >>= 1) { if (tid < t) red[tid] += red[tid + t]; __syncthreads(); }
  float inv = rsqrtf(red[0] * (1.f / Dn) + 1e-5f);
#pragma unroll
  for (int j = 0; j < 4; ++j) {
    int dcol = j * 256 + tid;
    float y = (v[j] - mean) * inv * g[dcol] + be[dcol];
    outF[row * Dn + dcol] = y;
    if (outB) outB[row * Dn + dcol] = (__bf16)y;
  }
}
__global__ __launch_bounds__(256)
void l2norm_rows_kernel(const float* __restrict__ inF, const __bf16* __restrict__ inB,
                        __bf16* __restrict__ outB) {
  const long row = blockIdx.x;
  const int tid = threadIdx.x;
  float v[4]; float ss = 0.f;
#pragma unroll
  for (int j = 0; j < 4; ++j) {
    long o = row * Dn + j * 256 + tid;
    float x = inF ? inF[o] : (float)inB[o];
    v[j] = x; ss += x * x;
  }
  __shared__ float red[256];
  red[tid] = ss; __syncthreads();
  for (int t = 128; t > 0; t >>= 1) { if (tid < t) red[tid] += red[tid + t]; __syncthreads(); }
  float inv = 1.f / (sqrtf(red[0]) + 1e-12f);
#pragma unroll
  for (int j = 0; j < 4; ++j) {
    long o = row * Dn + j * 256 + tid;
    outB[o] = (__bf16)(v[j] * inv);
  }
}
// opF[b,q,c] += (att[b,q,:] . mask[b,c,:]) / 3
__global__ void att_mask_accum_kernel(const float* __restrict__ att, const float* __restrict__ mask,
                                      float* __restrict__ opF) {
  long idx = (long)blockIdx.x * 256 + threadIdx.x;
  if (idx >= (long)Bn * Sn * 5) return;
  int c = (int)(idx % 5);
  int q = (int)((idx / 5) % Sn);
  int b = (int)(idx / (5 * Sn));
  const float* ar = att + ((long)b * Sn + q) * Sn;
  const float* mr = mask + ((long)b * 5 + c) * Sn;
  float s = 0.f;
  for (int k = 0; k < Sn; ++k) s += ar[k] * mr[k];
  opF[idx] += s * (1.f / 3.f);
}
// fuse_w [N,3K,3,3] f32 -> [tap][N][K] bf16
__global__ void pack_fuse_w_kernel(const float* __restrict__ w, __bf16* __restrict__ o) {
  long idx = (long)blockIdx.x * 256 + threadIdx.x;
  const long total = (long)9 * 1024 * 3072;
  if (idx >= total) return;
  int k = (int)(idx % 3072);
  long t = idx / 3072;
  int n = (int)(t % 1024);
  int tap = (int)(t / 1024);
  o[idx] = (__bf16)w[((long)n * 3072 + k) * 9 + tap];
}
// three [B,S,D] f32 maps -> padded NHWC [B][34][34][3072] bf16 (interior only; halo pre-zeroed)
__global__ void pack_feats_nhwc_kernel(const float* __restrict__ s0, const float* __restrict__ s1,
                                       const float* __restrict__ s2, __bf16* __restrict__ o) {
  long idx = (long)blockIdx.x * 256 + threadIdx.x;
  const long total = (long)Bn * Sn * 3072;
  if (idx >= total) return;
  int c = (int)(idx % 3072);
  long t = idx / 3072;
  int sp = (int)(t % Sn);
  int b = (int)(t / Sn);
  int y = sp >> 5, x = sp & 31;
  const float* sf = (c < 1024) ? s0 : (c < 2048 ? s1 : s2);
  int d = c & 1023;
  o[((long)(b * 34 + y + 1) * 34 + (x + 1)) * 3072 + c] =
      (__bf16)sf[((long)b * Sn + sp) * Dn + d];
}
// fused[m,ch] = relu((x + fuse_b)*bn_scale + bn_b)
__global__ void fuse_post_kernel(float* __restrict__ x, const float* __restrict__ cb,
                                 const float* __restrict__ g, const float* __restrict__ be) {
  long idx = (long)blockIdx.x * 256 + threadIdx.x;
  if (idx >= (long)Bn * Sn * Dn) return;
  int ch = (int)(idx & 1023);
  const float bnk = rsqrtf(1.f + 1e-5f);
  float v = (x[idx] + cb[ch]) * (g[ch] * bnk) + be[ch];
  x[idx] = fmaxf(v, 0.f);
}
// prediction conv: BN(concat(fused, op)) -> 3x3 conv to 5 channels
__global__ void pred_conv_kernel(const float* __restrict__ fused, const float* __restrict__ opF,
                                 const float* __restrict__ pg, const float* __restrict__ pb,
                                 const float* __restrict__ pw, const float* __restrict__ pbias,
                                 float* __restrict__ out) {
  int idx = blockIdx.x * 256 + threadIdx.x;   // B*5*1024
  if (idx >= Bn * 5 * Sn) return;
  int px = idx & 31, py = (idx >> 5) & 31;
  int oc = (idx >> 10) % 5, b = idx / (5 * Sn);
  const float bnk = rsqrtf(1.f + 1e-5f);
  float acc = pbias[oc];
  for (int ty = 0; ty < 3; ++ty) {
    int y = py + ty - 1; if ((unsigned)y >= 32u) continue;
    for (int tx = 0; tx < 3; ++tx) {
      int x = px + tx - 1; if ((unsigned)x >= 32u) continue;
      int s = y * 32 + x;
      const float* wrow = pw + (long)oc * 1029 * 9 + ty * 3 + tx;
      const float* fr = fused + ((long)b * Sn + s) * Dn;
      for (int c = 0; c < 1024; ++c) {
        float xv = fr[c] * (pg[c] * bnk) + pb[c];
        acc += xv * wrow[(long)c * 9];
      }
      const float* orow = opF + ((long)b * Sn + s) * 5;
      for (int c = 0; c < 5; ++c) {
        float xv = orow[c] * (pg[1024 + c] * bnk) + pb[1024 + c];
        acc += xv * wrow[(long)(1024 + c) * 9];
      }
    }
  }
  out[((long)b * 5 + oc) * Sn + py * 32 + px] = acc;
}

// ---------------- host orchestration ----------------
static inline dim3 ew(long n) { return dim3((unsigned)((n + 255) / 256)); }

extern "C" void kernel_launch(void* const* d_in, const int* in_sizes, int n_in,
                              void* d_out, int out_size, void* d_ws, size_t ws_size,
                              hipStream_t stream) {
  const float* qf    = (const float*)d_in[0];
  const float* rfim  = (const float*)d_in[1];
  const float* rmask = (const float*)d_in[2];
  const float* wq    = (const float*)d_in[3];
  const float* wk    = (const float*)d_in[4];
  const float* wv    = (const float*)d_in[5];
  const float* wo    = (const float*)d_in[6];
  const float* bo    = (const float*)d_in[7];
  const float* w1    = (const float*)d_in[8];
  const float* b1    = (const float*)d_in[9];
  const float* w2    = (const float*)d_in[10];
  const float* b2    = (const float*)d_in[11];
  const float* ln_g  = (const float*)d_in[12];
  const float* ln_b  = (const float*)d_in[13];
  const float* fin_g = (const float*)d_in[14];
  const float* fin_b = (const float*)d_in[15];
  const float* fw    = (const float*)d_in[16];
  const float* fb    = (const float*)d_in[17];
  const float* fbn_g = (const float*)d_in[18];
  const float* fbn_b = (const float*)d_in[19];
  const float* pbn_g = (const float*)d_in[20];
  const float* pbn_b = (const float*)d_in[21];
  const float* pw    = (const float*)d_in[22];
  const float* pb    = (const float*)d_in[23];
  float* out = (float*)d_out;

  // ---- workspace bump allocator ----
  char* wsp = (char*)d_ws;
  size_t off = 0;
  auto alloc = [&](size_t bytes) -> void* {
    void* p = wsp + off;
    off = (off + bytes + 255) & ~(size_t)255;
    return p;
  };
  const long BSD  = (long)Bn * Sn * Dn;
  const long BHLH = (long)BHn * Sn * HDn;
  __bf16* wqB = (__bf16*)alloc((size_t)NLn * HDn * HDn * 2);
  __bf16* wkB = (__bf16*)alloc((size_t)NLn * HDn * HDn * 2);
  __bf16* wvB = (__bf16*)alloc((size_t)NLn * HDn * HDn * 2);
  __bf16* woB = (__bf16*)alloc((size_t)NLn * Dn * Dn * 2);
  __bf16* w1B = (__bf16*)alloc((size_t)NLn * FFn * Dn * 2);
  __bf16* w2B = (__bf16*)alloc((size_t)NLn * Dn * FFn * 2);
  __bf16* fwB = (__bf16*)alloc((size_t)9 * 1024 * 3072 * 2);
  float*  srcF[3]; __bf16* srcB[3]; __bf16* supB[3]; float* sF[3];
  for (int i = 0; i < 3; ++i) srcF[i] = (float*)alloc(BSD * 4);
  for (int i = 0; i < 3; ++i) srcB[i] = (__bf16*)alloc(BSD * 2);
  for (int i = 0; i < 3; ++i) supB[i] = (__bf16*)alloc(BSD * 2);
  for (int i = 0; i < 3; ++i) sF[i]   = (float*)alloc(BSD * 4);
  float*  projF  = (float*)alloc(BSD * 4);
  __bf16* vB     = (__bf16*)alloc(BHLH * 2);
  __bf16* kB     = (__bf16*)alloc(BHLH * 2);
  __bf16* qB     = (__bf16*)alloc(BHLH * 2);
  float*  energyF= (float*)alloc((size_t)BHn * Sn * Sn * 4);
  __bf16* attB   = (__bf16*)alloc((size_t)BHn * Sn * Sn * 2);
  float*  ctxF   = (float*)alloc(BHLH * 4);
  __bf16* ctxB   = (__bf16*)alloc(BSD * 2);
  float*  outF   = (float*)alloc(BSD * 4);
  __bf16* outB   = (__bf16*)alloc(BSD * 2);
  float*  hidF   = (float*)alloc((size_t)Bn * Sn * FFn * 4);
  __bf16* hidB   = (__bf16*)alloc((size_t)Bn * Sn * FFn * 2);
  float*  t2F    = (float*)alloc(BSD * 4);
  __bf16* snB    = (__bf16*)alloc(BSD * 2);
  __bf16* spnB   = (__bf16*)alloc(BSD * 2);
  float*  opF    = (float*)alloc((size_t)Bn * Sn * 5 * 4);
  const long convInE = (long)Bn * 34 * 34 * 3072;   // padded NHWC
  __bf16* convIn = (__bf16*)alloc((size_t)convInE * 2);
  float*  fusedF = (float*)alloc(BSD * 4);
  (void)ws_size; (void)n_in; (void)in_sizes; (void)out_size;

  // ---- weight conversion ----
  f2b_kernel<<<ew((long)NLn*HDn*HDn), 256, 0, stream>>>(wq, wqB, (long)NLn*HDn*HDn);
  f2b_kernel<<<ew((long)NLn*HDn*HDn), 256, 0, stream>>>(wk, wkB, (long)NLn*HDn*HDn);
  f2b_kernel<<<ew((long)NLn*HDn*HDn), 256, 0, stream>>>(wv, wvB, (long)NLn*HDn*HDn);
  f2b_kernel<<<ew((long)NLn*Dn*Dn),  256, 0, stream>>>(wo, woB, (long)NLn*Dn*Dn);
  f2b_kernel<<<ew((long)NLn*FFn*Dn), 256, 0, stream>>>(w1, w1B, (long)NLn*FFn*Dn);
  f2b_kernel<<<ew((long)NLn*Dn*FFn), 256, 0, stream>>>(w2, w2B, (long)NLn*Dn*FFn);
  pack_fuse_w_kernel<<<ew((long)9*1024*3072), 256, 0, stream>>>(fw, fwB);

  // ---- inputs NCHW -> [B,S,D] ----
  for (int i = 0; i < 3; ++i) {
    transpose_feat_kernel<<<ew(BSD), 256, 0, stream>>>(qf + i * BSD, srcF[i], srcB[i]);
    transpose_feat_kernel<<<ew(BSD), 256, 0, stream>>>(rfim + i * BSD, nullptr, supB[i]);
  }

  const long sLH = (long)Sn * HDn, sLL = (long)Sn * Sn;
  // ---- 9 transformer layers ----
  for (int i = 0; i < NLn; ++i) {
    int li = i % 3;
    gemm_nt_kernel<<<dim3(1, 256, 1), 256, 0, stream>>>(supB[li], wvB + (long)i*HDn*HDn, projF,
        nullptr, Bn*Sn*Hn, HDn, HDn, 0, 0, 0, 0);
    repack_to_bh_kernel<<<ew(BSD), 256, 0, stream>>>(projF, vB);
    gemm_nt_kernel<<<dim3(1, 256, 1), 256, 0, stream>>>(supB[li], wkB + (long)i*HDn*HDn, projF,
        nullptr, Bn*Sn*Hn, HDn, HDn, 0, 0, 0, 0);
    repack_to_bh_kernel<<<ew(BSD), 256, 0, stream>>>(projF, kB);
    gemm_nt_kernel<<<dim3(1, 256, 1), 256, 0, stream>>>(srcB[li], wqB + (long)i*HDn*HDn, projF,
        nullptr, Bn*Sn*Hn, HDn, HDn, 0, 0, 0, 0);
    repack_to_bh_kernel<<<ew(BSD), 256, 0, stream>>>(projF, qB);
    gemm_nt_kernel<<<dim3(8, 8, BHn), 256, 0, stream>>>(qB, kB, energyF, nullptr,
        Sn, Sn, HDn, sLH, sLH, sLL, 0);
    softmax_rows_kernel<<<BHn * Sn, 256, 0, stream>>>(energyF, attB, Sn, 1.f / 32.f);
    gemm_nn_kernel<<<dim3(1, 8, BHn), 256, 0, stream>>>(attB, vB, ctxF,
        Sn, HDn, Sn, sLL, sLH, sLH);
    repack_from_bh_kernel<<<ew(BSD), 256, 0, stream>>>(ctxF, ctxB);
    gemm_nt_kernel<<<dim3(8, 32, 1), 256, 0, stream>>>(ctxB, woB + (long)i*Dn*Dn, outF,
        bo + (long)i*Dn, Bn*Sn, Dn, Dn, 0, 0, 0, 0);
    f2b_kernel<<<ew(BSD), 256, 0, stream>>>(outF, outB, BSD);
    gemm_nt_kernel<<<dim3(16, 32, 1), 256, 0, stream>>>(outB, w1B + (long)i*FFn*Dn, hidF,
        b1 + (long)i*FFn, Bn*Sn, FFn, Dn, 0, 0, 0, 1 /*relu*/);
    f2b_kernel<<<ew((long)Bn*Sn*FFn), 256, 0, stream>>>(hidF, hidB, (long)Bn*Sn*FFn);
    gemm_nt_kernel<<<dim3(8, 32, 1), 256, 0, stream>>>(hidB, w2B + (long)i*Dn*FFn, t2F,
        b2 + (long)i*Dn, Bn*Sn, Dn, FFn, 0, 0, 0, 0);
    ln_rows_kernel<<<Bn * Sn, 256, 0, stream>>>(outF, t2F, ln_g + (long)i*Dn, ln_b + (long)i*Dn,
        srcF[li], srcB[li]);
  }

  // ---- prediction heads ----
  zero_kernel<<<ew((long)Bn*Sn*5), 256, 0, stream>>>(opF, (long)Bn*Sn*5);
  for (int i = 0; i < 3; ++i) {
    add_nchw_kernel<<<ew(BSD), 256, 0, stream>>>(srcF[i], qf + i * BSD, outF);
    ln_rows_kernel<<<Bn * Sn, 256, 0, stream>>>(outF, nullptr, fin_g, fin_b, sF[i], nullptr);
    l2norm_rows_kernel<<<Bn * Sn, 256, 0, stream>>>(sF[i], nullptr, snB);
    l2norm_rows_kernel<<<Bn * Sn, 256, 0, stream>>>(nullptr, supB[i], spnB);
    gemm_nt_kernel<<<dim3(8, 8, Bn), 256, 0, stream>>>(snB, spnB, energyF, nullptr,
        Sn, Sn, Dn, (long)Sn*Dn, (long)Sn*Dn, sLL, 0);
    softmax_rows_kernel<<<Bn * Sn, 256, 0, stream>>>(energyF, nullptr, Sn, 1.f);
    att_mask_accum_kernel<<<ew((long)Bn*Sn*5), 256, 0, stream>>>(energyF, rmask, opF);
  }

  // ---- fuse conv: zero padded NHWC, pack, 9-tap implicit GEMM ----
  zero_kernel<<<ew(convInE / 2), 256, 0, stream>>>((float*)convIn, convInE / 2);
  pack_feats_nhwc_kernel<<<ew((long)Bn*Sn*3072), 256, 0, stream>>>(sF[0], sF[1], sF[2], convIn);
  for (int tap = 0; tap < 9; ++tap) {
    gemm_conv_tap_kernel<<<dim3(8, 32, 1), 256, 0, stream>>>(
        convIn, fwB + (long)tap * 1024 * 3072, fusedF, tap / 3 - 1, tap % 3 - 1, tap > 0);
  }
  fuse_post_kernel<<<ew(BSD), 256, 0, stream>>>(fusedF, fb, fbn_g, fbn_b);

  // ---- pred conv -> d_out ----
  pred_conv_kernel<<<ew((long)Bn*5*Sn), 256, 0, stream>>>(fusedF, opF, pbn_g, pbn_b, pw, pb, out);
}
